// SCNN_38628935860526
// MI455X (gfx1250) — matrix operand order repaired and checked by reading, not compiled, not run on hardware
//
#include <hip/hip_runtime.h>

// Problem dims (from reference)
#define C      128
#define Hh     128
#define Ww     256
#define Bsz    8
#define KTAPS  9
#define KCTOT  (KTAPS * C)        // 1152 = GEMM K for conv passes
#define KRES   576                // K kept resident in LDS (18 groups, 144 KB bf16)
#define RGRP   (KRES / 32)        // 18 resident 32-wide K groups
#define KCHUNK 192                // streamed K-chunk (48 KB per buffer)
#define NTHREADS 1024             // 32 wave32s -> one WGP per batch
#define NGRP_TOT   (KCTOT / 32)   // 36 K-groups per weight tensor
#define PKW_TENSOR (NGRP_TOT * 2 * C * 8)   // 73728 u32 of packed bf16 pairs per tensor

static_assert((KCTOT - KRES) % KCHUNK == 0, "streamed region is whole chunks");
// each wave issues exactly 3 async b128 ops per chunk stage and 9 per resident stage
static_assert(((KCHUNK / 2) * C / 4) % NTHREADS == 0, "uniform chunk issue count");
static_assert(((KRES   / 2) * C / 4) % NTHREADS == 0, "uniform resident issue count");

#define GAS __attribute__((address_space(1)))

typedef __attribute__((ext_vector_type(16))) __bf16 bf16x16;
typedef __attribute__((ext_vector_type(8)))  float  f32x8;
typedef __attribute__((ext_vector_type(4)))  unsigned u32x4;

__device__ __forceinline__ unsigned short f2bf_bits(float f) {
  union { __bf16 h; unsigned short s; } cv;
  cv.h = (__bf16)f;
  return cv.s;
}

__device__ __forceinline__ unsigned pack2_bf16(float lo, float hi) {
  union { __bf16 h[2]; unsigned u; } p;
  p.h[0] = (__bf16)lo;
  p.h[1] = (__bf16)hi;
  return p.u;
}

// ---------------------------------------------------------------------------
// Prep kernel: pack fp32 weights into the LDS-swizzled bf16-pair layout:
//   pk[ ((G*2 + h)*C + n)*8 + j ] = pack(W[G*32 + 16h + 2j][n], W[...+1][n])
// so staging is a contiguous async copy and each lane's B-frag is two
// ds_load_b128s.
// ---------------------------------------------------------------------------
__global__ void scnn_pack_weights(const float* __restrict__ Wd,
                                  const float* __restrict__ Wu,
                                  const float* __restrict__ Wl,
                                  unsigned* __restrict__ pk)
{
  const int idx = blockIdx.x * blockDim.x + threadIdx.x;
  if (idx >= 3 * PKW_TENSOR) return;
  const int t = idx / PKW_TENSOR;
  const int r = idx - t * PKW_TENSOR;
  const int j = r & 7;
  const int n = (r >> 3) & (C - 1);
  const int h = (r >> 10) & 1;
  const int G = r >> 11;
  const int k = G * 32 + 16 * h + 2 * j;
  GAS const float* W = (GAS const float*)(t == 0 ? Wd : (t == 1 ? Wu : Wl));
  ((GAS unsigned*)pk)[idx] = pack2_bf16(W[k * C + n], W[(k + 1) * C + n]);
}

// ---------------------------------------------------------------------------
// Issue staging of one K-range [k0, k0+clen) of the weight panel into LDS.
//  - packed path: contiguous async global->LDS copy (ASYNCcnt, no wait here)
//  - fallback:   synchronous fp32 load + cvt into the same swizzled layout
// ---------------------------------------------------------------------------
__device__ __forceinline__ void stage_issue(unsigned* dst,
                                            GAS const unsigned* WgPk,
                                            GAS const float* Wg,
                                            int k0, int clen, int tid)
{
  if (WgPk) {
    const int nvec = ((clen >> 1) * C) >> 2;           // u32x4 transfers
    GAS const unsigned* src = WgPk + (k0 >> 5) * (2 * C * 8);
    for (int i = tid; i < nvec; i += NTHREADS) {
      const unsigned ldsb = (unsigned)(size_t)(&dst[i * 4]);   // LDS byte offset
      const unsigned long long ga = (unsigned long long)(size_t)(src + i * 4);
      asm volatile("global_load_async_to_lds_b128 %0, %1, off"
                   :: "v"(ldsb), "v"(ga) : "memory");
    }
  } else {
    const int npairs = (clen >> 1) * C;
    for (int i = tid; i < npairs; i += NTHREADS) {
      const int j  = i & 7;
      const int n  = (i >> 3) & (C - 1);
      const int h  = (i >> 10) & 1;
      const int gl = i >> 11;
      const int k  = k0 + gl * 32 + 16 * h + 2 * j;
      dst[i] = pack2_bf16(Wg[k * C + n], Wg[(k + 1) * C + n]);
    }
  }
}

// Fill the resident weight region once per pass (K in [0, KRES)).
__device__ __forceinline__ void stage_resident(unsigned* sRes,
                                               GAS const unsigned* WgPk,
                                               GAS const float* Wg, int tid)
{
  stage_issue(sRes, WgPk, Wg, 0, KRES, tid);
  if (WgPk) asm volatile("s_wait_asynccnt 0x0" ::: "memory");
  __syncthreads();
}

// One WMMA K-group: A from carry slice (with conv tap shift), B from an LDS panel.
template<int NPOS, int TPW>
__device__ __forceinline__ void gemm_group(const u32x4* sA4, const u32x4* sB4,
                                           int gk, int gLocal, f32x8 (&acc)[TPW],
                                           int half, int l15, int nt, int mt0)
{
  const int t  = gk >> 7;                    // conv tap for this group
  const int cb = (gk & (C - 1)) + 16 * half;

  union { u32x4 x4[2]; bf16x16 v; } Bf;
  const int bb = (((gLocal * 2 + half) * C + nt * 16 + l15) << 3) >> 2;  // in u32x4
  Bf.x4[0] = sB4[bb];
  Bf.x4[1] = sB4[bb + 1];

  #pragma unroll
  for (int q = 0; q < TPW; ++q) {
    const int pos = (mt0 + 4 * q) * 16 + l15 + t - (KTAPS / 2);
    union { u32x4 x4[2]; bf16x16 v; } A;
    if (pos >= 0 && pos < NPOS) {
      const int a0 = (pos * C + cb) >> 3;    // u32x4 index (16B aligned)
      A.x4[0] = sA4[a0];
      A.x4[1] = sA4[a0 + 1];
    } else {
      const u32x4 z = {0u, 0u, 0u, 0u};
      A.x4[0] = z;
      A.x4[1] = z;
    }
    acc[q] = __builtin_amdgcn_wmma_f32_16x16x32_bf16(
        false, A.v, false, Bf.v, (short)0, acc[q], false, false);
  }
}

// ---------------------------------------------------------------------------
// One recurrence step (whole workgroup = one batch):
//   out[m][n] = add[m][n] + relu( sum_k prev[m+tap(k)-4][cin(k)] * W[kBase+k][n] + bias[n] )
// K = [kBase, kBase + kResGroups*32) comes from the resident LDS panel
// (zero barriers); the remaining kcStream is double-buffered via async copies.
// ---------------------------------------------------------------------------
template<int NPOS>
__device__ void conv_gemm_step(unsigned short* sA, const unsigned* sRes,
                               unsigned* sBw0, unsigned* sBw1, const float* sBias,
                               GAS const unsigned* WgPk, GAS const float* Wg,
                               int kBase, int kResGroups, int kcStream,
                               GAS const float* addSrc, int addStride,
                               GAS float* outDst, int outStride,
                               int tid)
{
  constexpr int TPW = (NPOS / 16) * 8 / 32;   // 16x16 tiles per wave (4 or 2)
  const int lane = tid & 31;
  const int wave = tid >> 5;
  const int half = lane >> 4;
  const int l15  = lane & 15;
  const int nt   = wave & 7;                  // N-tile constant per wave -> B-frag reuse
  const int mt0  = wave >> 3;
  const int kStreamBase = kBase + (kResGroups << 5);
  const int nStream = (kcStream + KCHUNK - 1) / KCHUNK;

  // Warm the residual-add row/column for the epilogue (one 128B line per thread).
  {
    const int e = tid * 32;
    if (e < NPOS * C)
      __builtin_prefetch((const void*)&addSrc[(size_t)(e >> 7) * addStride + (e & (C - 1))], 0, 3);
  }

  // Issue streamed chunk 0 early; it lands while the resident phase computes.
  {
    const int clen0 = kcStream < KCHUNK ? kcStream : KCHUNK;
    stage_issue(sBw0, WgPk, Wg, kStreamBase, clen0, tid);
  }

  f32x8 acc[TPW];
  #pragma unroll
  for (int q = 0; q < TPW; ++q) {
    #pragma unroll
    for (int r = 0; r < 8; ++r) acc[q][r] = 0.0f;
  }

  const u32x4* sA4 = (const u32x4*)sA;

  // ---- resident phase: no barriers, no waits ----
  const u32x4* sR4 = (const u32x4*)sRes;
  for (int g = 0; g < kResGroups; ++g)
    gemm_group<NPOS, TPW>(sA4, sR4, kBase + (g << 5), g, acc, half, l15, nt, mt0);

  // ---- streamed phase: double-buffered async chunks ----
  for (int sc = 0; sc < nStream; ++sc) {
    unsigned* cur = (sc & 1) ? sBw1 : sBw0;
    unsigned* nxt = (sc & 1) ? sBw0 : sBw1;
    const int k0   = kStreamBase + sc * KCHUNK;
    const int rem  = kcStream - sc * KCHUNK;
    const int clen = rem < KCHUNK ? rem : KCHUNK;

    if (sc + 1 < nStream) {
      const int remN  = rem - KCHUNK;
      const int clenN = remN < KCHUNK ? remN : KCHUNK;
      stage_issue(nxt, WgPk, Wg, k0 + KCHUNK, clenN, tid);    // overlap with GEMM
      if (WgPk) asm volatile("s_wait_asynccnt 0x3" ::: "memory");  // chunk sc resident
    } else {
      if (WgPk) asm volatile("s_wait_asynccnt 0x0" ::: "memory");
    }
    __syncthreads();                          // chunk sc visible to all waves

    const u32x4* sB4 = (const u32x4*)cur;
    const int ngroups = clen >> 5;
    for (int g = 0; g < ngroups; ++g)
      gemm_group<NPOS, TPW>(sA4, sB4, k0 + (g << 5), g, acc, half, l15, nt, mt0);

    __syncthreads();   // all waves done reading `cur` before it is re-staged
  }

  // Epilogue: bias + ReLU + residual add; write fp32 global, commit bf16 carry.
  // (trailing barrier above also guarantees all reads of sA are complete)
  #pragma unroll
  for (int q = 0; q < TPW; ++q) {
    const int mt = mt0 + 4 * q;
    const int n  = nt * 16 + l15;
    const float bias = sBias[n];
    #pragma unroll
    for (int r = 0; r < 8; ++r) {
      const int m = mt * 16 + r + 8 * half;   // f32 D layout: M = r + 8*(lane>=16)
      float v = acc[q][r] + bias;
      v = v > 0.0f ? v : 0.0f;
      v += addSrc[(size_t)m * addStride + n];
      outDst[(size_t)m * outStride + n] = v;
      sA[m * C + n] = f2bf_bits(v);
    }
  }
  __threadfence_block();
  __syncthreads();
}

__global__ __launch_bounds__(NTHREADS, 1)
void scnn_persistent(const float* __restrict__ x,
                     const float* __restrict__ Wd, const float* __restrict__ bd,
                     const float* __restrict__ Wu, const float* __restrict__ bu,
                     const float* __restrict__ Wl, const float* __restrict__ bl,
                     float* __restrict__ z, const unsigned* wsPk)
{
  __shared__ __align__(16) unsigned short sA[Ww * C];           // 64 KB carry (bf16)
  __shared__ __align__(16) unsigned sRes[(KRES / 2) * C];       // 144 KB resident weights
  __shared__ __align__(16) unsigned sBw0[(KCHUNK / 2) * C];     // 48 KB stream buf 0
  __shared__ __align__(16) unsigned sBw1[(KCHUNK / 2) * C];     // 48 KB stream buf 1
  __shared__ float sBias[C];

  const int b   = blockIdx.x;                                   // one batch per WGP
  const int tid = threadIdx.x;

  GAS const float* gx  = (GAS const float*)x + (size_t)b * Hh * Ww * C;
  GAS float*       gz  = (GAS float*)z + (size_t)b * Hh * Ww * C;
  GAS const float* gWd = (GAS const float*)Wd;
  GAS const float* gWu = (GAS const float*)Wu;
  GAS const float* gWl = (GAS const float*)Wl;
  GAS const float* gbd = (GAS const float*)bd;
  GAS const float* gbu = (GAS const float*)bu;
  GAS const float* gbl = (GAS const float*)bl;
  GAS const unsigned* pkd = wsPk ? (GAS const unsigned*)wsPk : (GAS const unsigned*)0;
  GAS const unsigned* pku = pkd ? pkd + PKW_TENSOR : pkd;
  GAS const unsigned* pkl = pkd ? pkd + 2 * PKW_TENSOR : pkd;

  // ---------------- down pass: z[i] = x[i] + relu(conv_W(z[i-1]; Wd) + bd) ----------------
  for (int i = tid; i < Ww * C; i += NTHREADS) {                // z[0] = x[0]
    const float v = gx[i];
    gz[i] = v;
    sA[i] = f2bf_bits(v);
  }
  if (tid < C) sBias[tid] = gbd[tid];
  __threadfence_block();
  __syncthreads();
  stage_resident(sRes, pkd, gWd, tid);

  for (int i = 1; i < Hh; ++i) {
    GAS const float* add = gx + (size_t)i * Ww * C;
    GAS float*       out = gz + (size_t)i * Ww * C;
    conv_gemm_step<Ww>(sA, sRes, sBw0, sBw1, sBias, pkd, gWd,
                       0, RGRP, KCTOT - KRES, add, C, out, C, tid);
  }

  // ---------------- up pass: z[i] += relu(conv_W(z[i+1]; Wu) + bu), i = H-2..1 ----------------
  for (int i = tid; i < Ww * C; i += NTHREADS)
    sA[i] = f2bf_bits(gz[(size_t)(Hh - 1) * Ww * C + i]);
  if (tid < C) sBias[tid] = gbu[tid];
  __threadfence_block();
  __syncthreads();
  stage_resident(sRes, pku, gWu, tid);

  for (int i = Hh - 2; i >= 1; --i) {
    GAS float* row = gz + (size_t)i * Ww * C;
    conv_gemm_step<Ww>(sA, sRes, sBw0, sBw1, sBias, pku, gWu,
                       0, RGRP, KCTOT - KRES, row, C, row, C, tid);
  }

  // ---------------- left pass: z[:, :, j] += relu(conv_H(z[:, :, j-1]; Wl) + bl) ----------------
  for (int i = tid; i < Hh * C; i += NTHREADS) {
    const int h = i >> 7, c = i & (C - 1);
    sA[i] = f2bf_bits(gz[(size_t)h * Ww * C + c]);              // column 0
  }
  if (tid < C) sBias[tid] = gbl[tid];
  __threadfence_block();
  __syncthreads();
  stage_resident(sRes, pkl, gWl, tid);

  for (int j = 1; j < Ww; ++j) {
    GAS float* col = gz + (size_t)j * C;
    conv_gemm_step<Hh>(sA, sRes, sBw0, sBw1, sBias, pkl, gWl,
                       0, RGRP, KCTOT - KRES, col, Ww * C, col, Ww * C, tid);
  }

  // ---------------- right pass: center-tap 1x1 with Wd[4], j = W-2..1 ----------------
  for (int i = tid; i < Hh * C; i += NTHREADS) {
    const int h = i >> 7, c = i & (C - 1);
    sA[i] = f2bf_bits(gz[(size_t)h * Ww * C + (size_t)(Ww - 1) * C + c]);
  }
  if (tid < C) sBias[tid] = gbd[tid];
  __threadfence_block();
  __syncthreads();

  for (int j = Ww - 2; j >= 1; --j) {
    GAS float* col = gz + (size_t)j * C;
    // K=128 at kBase=512 (center tap): no resident region, single streamed chunk
    conv_gemm_step<Hh>(sA, sRes, sBw0, sBw1, sBias, pkd, gWd,
                       (KTAPS / 2) * C, 0, C, col, Ww * C, col, Ww * C, tid);
  }
}

extern "C" void kernel_launch(void* const* d_in, const int* in_sizes, int n_in,
                              void* d_out, int out_size, void* d_ws, size_t ws_size,
                              hipStream_t stream) {
  (void)in_sizes; (void)n_in; (void)out_size;
  const float* x  = (const float*)d_in[0];
  const float* Wd = (const float*)d_in[1];
  const float* bd = (const float*)d_in[2];
  const float* Wu = (const float*)d_in[3];
  const float* bu = (const float*)d_in[4];
  const float* Wl = (const float*)d_in[5];
  const float* bl = (const float*)d_in[6];
  float* z = (float*)d_out;

  const size_t pkBytes = (size_t)3 * PKW_TENSOR * sizeof(unsigned);
  const bool useWs = (ws_size >= pkBytes);
  const unsigned* pk = useWs ? (const unsigned*)d_ws : nullptr;

  if (useWs) {
    const int total = 3 * PKW_TENSOR;
    hipLaunchKernelGGL(scnn_pack_weights, dim3((total + 255) / 256), dim3(256), 0, stream,
                       Wd, Wu, Wl, (unsigned*)d_ws);
  }
  hipLaunchKernelGGL(scnn_persistent, dim3(Bsz), dim3(NTHREADS), 0, stream,
                     x, Wd, bd, Wu, bu, Wl, bl, z, pk);
}
